// DCRNN_87162066305587
// MI455X (gfx1250) — matrix operand order repaired and checked by reading, not compiled, not run on hardware
//
#include <hip/hip_runtime.h>
#include <hip/hip_bf16.h>

typedef __attribute__((ext_vector_type(16))) _Float16 v16h;
typedef __attribute__((ext_vector_type(8)))  float    v8f;

// Problem constants (match reference).
constexpr int B  = 4;
constexpr int N  = 5000;
constexpr int SL = 128;     // SEQ_LEN
constexpr int OL = 128;     // OUT_LEN
constexpr int E  = 80000;
constexpr int MT = (N + 15) / 16;   // 313 row tiles
constexpr int CT = OL / 16;         // 8 column tiles per gate

// ---------------------------------------------------------------------------
// 1) zero a float4 region
__global__ __launch_bounds__(256) void k_zero4(float4* __restrict__ p, long n4) {
    long t = (long)blockIdx.x * blockDim.x + threadIdx.x;
    if (t < n4) p[t] = make_float4(0.f, 0.f, 0.f, 0.f);
}

// 2) weighted out-/in-degrees
__global__ __launch_bounds__(256) void k_deg(const int* __restrict__ ei,
                                             const float* __restrict__ w,
                                             float* __restrict__ degO,
                                             float* __restrict__ degI) {
    int e = blockIdx.x * blockDim.x + threadIdx.x;
    if (e >= E) return;
    atomicAdd(degO + ei[e],     w[e]);   // src
    atomicAdd(degI + ei[E + e], w[e]);   // dst
}

// 3) per-edge coefficients with safe reciprocal
__global__ __launch_bounds__(256) void k_coef(const int* __restrict__ ei,
                                              const float* __restrict__ w,
                                              const float* __restrict__ degO,
                                              const float* __restrict__ degI,
                                              float* __restrict__ co,
                                              float* __restrict__ ci) {
    int e = blockIdx.x * blockDim.x + threadIdx.x;
    if (e >= E) return;
    float doo = degO[ei[e]];
    float dii = degI[ei[E + e]];
    co[e] = (doo > 0.f) ? w[e] / doo : 0.f;
    ci[e] = (dii > 0.f) ? w[e] / dii : 0.f;
}

// 4) Build transposed effective f16 weights Wt[src][cidx][k]:
//    cidx < 128 -> z-gate column, cidx >= 128 -> h-gate column.
//    src 0: W[0,0]+W[1,0]; src 1: W[0,1]; src 2: W[1,1] (first 128 rows only;
//    the H0 half is multiplied by zeros in the reference).
__global__ __launch_bounds__(256) void k_prepw(const float* __restrict__ Wz,
                                               const float* __restrict__ Wh,
                                               _Float16* __restrict__ Wt) {
    int tid = blockIdx.x * blockDim.x + threadIdx.x;   // 3*256*128
    if (tid >= 3 * 256 * 128) return;
    int k    = tid & 127;
    int cidx = (tid >> 7) & 255;
    int s    = tid >> 15;
    const float* W = (cidx < 128) ? Wz : Wh;
    int c = cidx & 127;
    // W shape (2,2,256,128): slab (k1,k2) at (k1*2+k2)*256*128, row k, col c.
    float v;
    if (s == 0)      v = W[0 * 32768 + k * 128 + c] + W[2 * 32768 + k * 128 + c];
    else if (s == 1) v = W[1 * 32768 + k * 128 + c];
    else             v = W[3 * 32768 + k * 128 + c];
    Wt[((size_t)s * 256 + cidx) * 128 + k] = (_Float16)v;
}

// 5) SpMM scatter: Txo[b][dst] += co[e]*X[b][src]; Txi[b][src] += ci[e]*X[b][dst]
//    one thread per (batch, edge, 4-column group)
__global__ __launch_bounds__(256) void k_scatter(const float* __restrict__ X,
                                                 const int* __restrict__ ei,
                                                 const float* __restrict__ co,
                                                 const float* __restrict__ ci,
                                                 float* __restrict__ Txo,
                                                 float* __restrict__ Txi) {
    long tid = (long)blockIdx.x * blockDim.x + threadIdx.x;
    const long TOT = (long)B * E * 32;
    if (tid >= TOT) return;
    int  c4 = (int)(tid & 31);
    long t  = tid >> 5;
    int  e  = (int)(t % E);
    int  b  = (int)(t / E);
    int  s  = ei[e];
    int  d  = ei[E + e];
    float wo = co[e], wi = ci[e];
    const float* xs = X   + ((size_t)b * N + s) * 128 + c4 * 4;
    const float* xd = X   + ((size_t)b * N + d) * 128 + c4 * 4;
    float*       to = Txo + ((size_t)b * N + d) * 128 + c4 * 4;
    float*       ti = Txi + ((size_t)b * N + s) * 128 + c4 * 4;
    float4 vs = *(const float4*)xs;
    float4 vd = *(const float4*)xd;
    atomicAdd(to + 0, wo * vs.x); atomicAdd(to + 1, wo * vs.y);
    atomicAdd(to + 2, wo * vs.z); atomicAdd(to + 3, wo * vs.w);
    atomicAdd(ti + 0, wi * vd.x); atomicAdd(ti + 1, wi * vd.y);
    atomicAdd(ti + 2, wi * vd.z); atomicAdd(ti + 3, wi * vd.w);
}

// 6) Fused WMMA GEMM + epilogue.
//    One wave owns a 16x16 output tile and accumulates BOTH gates:
//      cz += A(s) * Wt[s][col_z]   ch += A(s) * Wt[s][col_z+128]   s = X,Txo,Txi
//    then out = sigmoid(-(cz+bz)) * tanh(ch+bh).
//    Fragment layouts: 16-bit A 16x32 / B 32x16 in contiguous K order (k = 16*hi + e), f32 C 16x16.
__global__ __launch_bounds__(128) void k_gemm(const float* __restrict__ X,
                                              const float* __restrict__ Txo,
                                              const float* __restrict__ Txi,
                                              const _Float16* __restrict__ Wt,
                                              const float* __restrict__ bz,
                                              const float* __restrict__ bh,
                                              float* __restrict__ out) {
    const int lane = threadIdx.x & 31;
    const int wave = threadIdx.x >> 5;
    int gw = blockIdx.x * 4 + wave;                  // wave-uniform
    const int TOT = B * CT * MT;
    if (gw >= TOT) return;                           // uniform exit: EXEC stays all-ones
    int mtile = gw % MT;
    int t     = gw / MT;
    int ctile = t & (CT - 1);
    int b     = t / CT;

    const int ml  = lane & 15;
    const int hi  = lane >> 4;
    const int klo = hi * 16;    // A: lanes<16 get K 0..15, lanes>=16 get 16..31
    const int kb  = hi * 16;    // B: lanes<16 get K 0..15, lanes>=16 get 16..31

    int rowa = mtile * 16 + ml;
    if (rowa > N - 1) rowa = N - 1;                  // clamp; out-of-range rows never stored

    const float* Xb = X   + (size_t)b * N * 128;
    const float* Ob = Txo + (size_t)b * N * 128;
    const float* Ib = Txi + (size_t)b * N * 128;
    const int colz = ctile * 16 + ml;                // this lane's output column (z gate)

    v8f cz = {};
    v8f ch = {};

#pragma unroll
    for (int s = 0; s < 3; ++s) {
        const float* S = (s == 0) ? Xb : ((s == 1) ? Ob : Ib);
        const float* prow = S + (size_t)rowa * 128;
        const _Float16* wrow = Wt + ((size_t)(s * 256 + colz)) * 128;
#pragma unroll
        for (int kk = 0; kk < 4; ++kk) {             // K = 128 in steps of 32
            const float* p = prow + kk * 32 + klo;
            float4 f0 = *(const float4*)(p);
            float4 f1 = *(const float4*)(p + 4);
            float4 f2 = *(const float4*)(p + 8);
            float4 f3 = *(const float4*)(p + 12);
            float tmp[16] = {f0.x, f0.y, f0.z, f0.w, f1.x, f1.y, f1.z, f1.w,
                             f2.x, f2.y, f2.z, f2.w, f3.x, f3.y, f3.z, f3.w};
            v16h a;
#pragma unroll
            for (int e2 = 0; e2 < 16; ++e2) a[e2] = (_Float16)tmp[e2];

            const _Float16* q = wrow + kk * 32 + kb; // 16 contiguous halves (32B)
            v16h bw_z = *(const v16h*)q;
            v16h bw_h = *(const v16h*)(q + 128 * 128); // h-gate columns +128 rows
            cz = __builtin_amdgcn_wmma_f32_16x16x32_f16(false, a, false, bw_z,
                                                        (short)0, cz, false, false);
            ch = __builtin_amdgcn_wmma_f32_16x16x32_f16(false, a, false, bw_h,
                                                        (short)0, ch, false, false);
        }
    }

    // epilogue: C layout — VGPR r holds row r (lanes<16) / r+8 (lanes>=16), col = lane%16
    float bzs = bz[colz];
    float bhs = bh[colz];
    int rbase = mtile * 16 + hi * 8;
    float* ob = out + (size_t)b * N * 128;
#pragma unroll
    for (int r = 0; r < 8; ++r) {
        int row = rbase + r;
        if (row < N) {
            float zv = cz[r] + bzs;
            float hv = ch[r] + bhs;
            float g  = 1.0f / (1.0f + __expf(zv));   // 1 - sigmoid(zv) = sigmoid(-zv)
            ob[(size_t)row * 128 + colz] = g * tanhf(hv);
        }
    }
}

// ---------------------------------------------------------------------------
extern "C" void kernel_launch(void* const* d_in, const int* in_sizes, int n_in,
                              void* d_out, int out_size, void* d_ws, size_t ws_size,
                              hipStream_t stream) {
    (void)in_sizes; (void)n_in; (void)out_size; (void)ws_size;

    const float* x  = (const float*)d_in[0];
    const int*   ei = (const int*)d_in[1];
    const float* ew = (const float*)d_in[2];
    const float* Wz = (const float*)d_in[3];
    const float* bz = (const float*)d_in[4];
    // d_in[5] (Wr), d_in[6] (br): provably dead in the reference (H0 == 0)
    const float* Wh = (const float*)d_in[7];
    const float* bh = (const float*)d_in[8];
    float* out = (float*)d_out;

    // workspace carve-up (all offsets 32B-aligned)
    float* degO = (float*)d_ws;                          // N
    float* degI = degO + N;                              // N
    float* co   = degI + N;                              // E
    float* ci   = co + E;                                // E
    float* txo  = ci + E;                                // B*N*128
    float* txi  = txo + (size_t)B * N * 128;             // B*N*128
    _Float16* Wt = (_Float16*)(txi + (size_t)B * N * 128); // 3*256*128 halves

    // 1) zero accumulators
    long n4a = (2L * N) / 4;
    k_zero4<<<(int)((n4a + 255) / 256), 256, 0, stream>>>((float4*)degO, n4a);
    long n4b = (2L * B * N * 128) / 4;
    k_zero4<<<(int)((n4b + 255) / 256), 256, 0, stream>>>((float4*)txo, n4b);

    // 2-4) degrees, coefficients, packed f16 weights
    k_deg  <<<(E + 255) / 256, 256, 0, stream>>>(ei, ew, degO, degI);
    k_coef <<<(E + 255) / 256, 256, 0, stream>>>(ei, ew, degO, degI, co, ci);
    k_prepw<<<(3 * 256 * 128 + 255) / 256, 256, 0, stream>>>(Wz, Wh, Wt);

    // 5) sparse diffusion aggregation
    long ts = (long)B * E * 32;
    k_scatter<<<(int)((ts + 255) / 256), 256, 0, stream>>>(x, ei, co, ci, txo, txi);

    // 6) fused WMMA GEMM + gate epilogue (4 waves / block)
    int waves  = B * CT * MT;          // 10016
    int blocks = (waves + 3) / 4;      // 2504
    k_gemm<<<blocks, 128, 0, stream>>>(x, txo, txi, Wt, bz, bh, out);
}
